// MLP_CIM_42116449304871
// MI455X (gfx1250) — compile-verified, matrix operand order checked by reading, not run on hardware
//
#include <hip/hip_runtime.h>

// Binarized MLP inference, fused for MI455X (gfx1250, wave32, WMMA).
// Layer1: x[131072,784] (f32->f16) @ signW1[784,512] via v_wmma_f32_16x16x32_f16
// Layer2: sign(+-1 f16) [.,512] @ signW2[512,32] via WMMA (exact in f16/f32)
// Layer3: tiny [.,32]x[32,10] in VALU.
// BN+hardtanh+binarize collapses to "acc >= thresh ? +1 : -1".

typedef __attribute__((ext_vector_type(16))) _Float16 v16h;
typedef __attribute__((ext_vector_type(8)))  float    v8f;
typedef __attribute__((ext_vector_type(4)))  float    f32x4;

#define LDS_AS __attribute__((address_space(3)))

// ---- CDNA5 async copy: global -> LDS, 16 bytes per lane, tracked by ASYNCcnt.
static __device__ __forceinline__ void async_b128(const void* gptr, void* lptr) {
  unsigned loff = (unsigned)(unsigned long long)(LDS_AS char*)lptr; // LDS byte address
  unsigned long long g = (unsigned long long)gptr;
  asm volatile("global_load_async_to_lds_b128 %0, %1, off"
               :: "v"(loff), "v"(g) : "memory");
}

// Copy one 32KB W1 fragment slab (256 threads x 8 x 16B).
static __device__ __forceinline__ void slab_prefetch(const _Float16* gsrc, char* ldst, int tid) {
  const char* gc = (const char*)gsrc;
#pragma unroll
  for (int i = 0; i < 8; ++i) {
    const int off = tid * 16 + i * 4096;
    async_b128(gc + off, ldst + off);
  }
}

// ---------------------------------------------------------------------------
// Prep: binarize weights into WMMA-B fragment order (f16 +-1), thresholds.
// B-frag layout (16x16x32 f16): lane l holds col N = nt*16 + (l&15), group
// g = l>>4; half h (0..15) holds K = kc*32 + koff(h) + 8*g, koff = h<8 ? h : h+8.
// Stored so each lane's 16 halves are contiguous 32B: [kc][nt][lane][16h].
// ---------------------------------------------------------------------------
__global__ __launch_bounds__(256) void bnn_prep(
    const float* __restrict__ W1, const float* __restrict__ b1,
    const float* __restrict__ g1, const float* __restrict__ be1,
    const float* __restrict__ m1, const float* __restrict__ v1,
    const float* __restrict__ W2, const float* __restrict__ b2,
    const float* __restrict__ g2, const float* __restrict__ be2,
    const float* __restrict__ m2, const float* __restrict__ v2,
    const float* __restrict__ W3,
    _Float16* __restrict__ w1frag, _Float16* __restrict__ w2frag,
    float* __restrict__ th1, float* __restrict__ th2, float* __restrict__ w3b)
{
  int idx = blockIdx.x * 256 + threadIdx.x;

  if (idx < 409600) {                      // w1frag: 25 chunks x 32 nt x 32 lanes x 16 halves
    const int h    = idx & 15;
    const int lane = (idx >> 4) & 31;
    const int nt   = (idx >> 9) & 31;
    const int c    = idx >> 14;            // 0..24
    const int n    = nt * 16 + (lane & 15);
    const int koff = (h < 8) ? h : h + 8;
    const int k    = c * 32 + koff + 8 * (lane >> 4);
    _Float16 v = (_Float16)0.0f;           // zero-pad K in [784,800)
    if (k < 784) v = (W1[n * 784 + k] >= 0.0f) ? (_Float16)1.0f : (_Float16)-1.0f;
    w1frag[idx] = v;
    return;
  }
  idx -= 409600;
  if (idx < 16384) {                       // w2frag: 16 chunks x 2 nt x 32 lanes x 16 halves
    const int h    = idx & 15;
    const int lane = (idx >> 4) & 31;
    const int nt   = (idx >> 9) & 1;
    const int c    = idx >> 10;            // 0..15
    const int n    = nt * 16 + (lane & 15);
    const int koff = (h < 8) ? h : h + 8;
    const int k    = c * 32 + koff + 8 * (lane >> 4);
    w2frag[idx] = (W2[n * 512 + k] >= 0.0f) ? (_Float16)1.0f : (_Float16)-1.0f;
    return;
  }
  idx -= 16384;
  if (idx < 512) {                         // layer-1 sign threshold (folds b1, BN)
    const float a = g1[idx] * rsqrtf(v1[idx] + 1e-5f);
    th1[idx] = m1[idx] - b1[idx] - be1[idx] / a;
    return;
  }
  idx -= 512;
  if (idx < 32) {                          // layer-2 sign threshold
    const float a = g2[idx] * rsqrtf(v2[idx] + 1e-5f);
    th2[idx] = m2[idx] - b2[idx] - be2[idx] / a;
    return;
  }
  idx -= 32;
  if (idx < 320) {                         // binarized W3, f32
    w3b[idx] = (W3[idx] >= 0.0f) ? 1.0f : -1.0f;
  }
}

// ---------------------------------------------------------------------------
// Fused MLP. Block = 64 rows x full width. 8 waves: layer1 wave = (mt = w&3
// row-tile of 16, nh = w>>2 N-half of 256). 16 v8f accumulators per wave.
// ---------------------------------------------------------------------------
__global__ __launch_bounds__(256) void bnn_fused(
    const float* __restrict__ x,
    const _Float16* __restrict__ w1frag,
    const _Float16* __restrict__ w2frag,
    const float* __restrict__ th1,
    const float* __restrict__ th2,
    const float* __restrict__ w3b,
    const float* __restrict__ b3,
    float* __restrict__ out)
{
  __shared__ __align__(64) char smem[65536 + 8192];
  char*  slabs = smem;                     // 2 x 32KB W1 slabs; later aliased by s1 A-frags (64KB)
  float* s2    = (float*)(smem + 65536);   // sign(h2): 64 x 32 f32

  const int tid  = threadIdx.x;
  const int lane = tid & 31;
  const int w    = tid >> 5;
  const int l15  = lane & 15;
  const int g    = lane >> 4;
  const int mt   = w & 3;                  // row tile (16 rows)
  const int nh   = w >> 2;                 // N half (256 cols)
  const int rowBase = blockIdx.x * 64;
  const float* xr = x + (size_t)(rowBase + mt * 16 + l15) * 784;

  v8f acc[16] = {};

  // Prime the pipeline: slab 0 -> buffer 0 (async, ASYNCcnt-tracked).
  slab_prefetch(w1frag, slabs, tid);

  for (int kc = 0; kc < 25; ++kc) {
    char* slab  = slabs + ((kc & 1) ? 32768 : 0);
    char* nslab = slabs + ((kc & 1) ? 0 : 32768);
    if (kc + 1 < 25) {
      // Prefetch next slab, then wait until only those 8 newest async ops
      // remain outstanding (async-to-LDS completes in order) => slab kc ready.
      slab_prefetch(w1frag + (kc + 1) * 16384, nslab, tid);
      asm volatile("s_wait_asynccnt 0x8" ::: "memory");
    } else {
      asm volatile("s_wait_asynccnt 0x0" ::: "memory");
    }
    __syncthreads();

    // ---- A fragment: x rows in f32 -> f16, per ISA 16-bit A layout:
    // lane (M=l15, g), halves 0..7 = K kb+8g..+7, halves 8..15 = K kb+16+8g..+7
    const int kb = kc * 32;
    const float* xs = xr + kb + g * 8;
    const f32x4 s0a = *(const f32x4*)(xs);
    const f32x4 s0b = *(const f32x4*)(xs + 4);
    f32x4 s1a = {0.f, 0.f, 0.f, 0.f};
    f32x4 s1b = {0.f, 0.f, 0.f, 0.f};
    if (kb + 16 + g * 8 + 8 <= 784) {      // uniform: only false for kb==768
      s1a = *(const f32x4*)(xs + 16);
      s1b = *(const f32x4*)(xs + 20);
    }
    v16h A;
#pragma unroll
    for (int i = 0; i < 4; ++i) {
      A[i]      = (_Float16)s0a[i];
      A[4 + i]  = (_Float16)s0b[i];
      A[8 + i]  = (_Float16)s1a[i];
      A[12 + i] = (_Float16)s1b[i];
    }

#pragma unroll
    for (int nt = 0; nt < 16; ++nt) {
      const v16h B = *(const v16h*)(slab + (((nh * 16 + nt) * 32 + lane) << 5));
      acc[nt] = __builtin_amdgcn_wmma_f32_16x16x32_f16(
          false, A, false, B, (short)0, acc[nt], false, false);
    }
    __syncthreads();                        // all reads of this slab done before overwrite
  }

  // ---- Layer-1 epilogue: acc >= th ? +1 : -1, scattered into the layer-2
  // A-fragment layout (aliases the slab buffers, 64KB = 16kc x 4mt x 32lane x 32B).
  _Float16* s1frag = (_Float16*)slabs;
#pragma unroll
  for (int nt = 0; nt < 16; ++nt) {
    const int N   = nh * 256 + nt * 16 + l15;    // layer-1 col == layer-2 K
    const float th = th1[N];
    const int kc2 = N >> 5;
    const int kk  = N & 31;
    const int g2  = (kk >> 3) & 1;
    const int q   = kk - 8 * g2;
    const int h   = (q < 8) ? q : q - 8;
    const int fragBase = ((kc2 * 4 + mt) * 32) * 16 + h;   // halves
#pragma unroll
    for (int r = 0; r < 8; ++r) {
      const int m     = r + 8 * g;               // C-layout row within 16x16 tile
      const int lane2 = m + 16 * g2;
      s1frag[fragBase + lane2 * 16] =
          (acc[nt][r] >= th) ? (_Float16)1.0f : (_Float16)-1.0f;
    }
  }
  __syncthreads();

  // ---- Layer 2: [64 x 512] (+-1 f16) x [512 x 32] (+-1 f16), exact in f32.
  const int mt2 = w & 3;
  const int nt2 = w >> 2;                        // 0..1
  v8f acc2 = {};
#pragma unroll
  for (int kc2 = 0; kc2 < 16; ++kc2) {
    const v16h A2 = *(const v16h*)(slabs + (((kc2 * 4 + mt2) * 32 + lane) << 5));
    const v16h B2 = *(const v16h*)(w2frag + (((kc2 * 2 + nt2) * 32 + lane) << 4));
    acc2 = __builtin_amdgcn_wmma_f32_16x16x32_f16(
        false, A2, false, B2, (short)0, acc2, false, false);
  }
  {
    const int N2 = nt2 * 16 + l15;
    const float th = th2[N2];
#pragma unroll
    for (int r = 0; r < 8; ++r) {
      const int Mloc = mt2 * 16 + r + 8 * g;
      s2[Mloc * 32 + N2] = (acc2[r] >= th) ? 1.0f : -1.0f;
    }
  }
  __syncthreads();

  // ---- Layer 3: out[64 x 10] = s2[64 x 32] @ w3b^T + b3 (tiny, VALU).
  for (int idx = tid; idx < 640; idx += 256) {
    const int rrow = idx / 10;
    const int j    = idx - rrow * 10;
    float sum = b3[j];
    const float* sr = s2 + rrow * 32;
    const float* wj = w3b + j * 32;
#pragma unroll
    for (int k = 0; k < 32; ++k) sum += sr[k] * wj[k];
    out[(size_t)(rowBase + rrow) * 10 + j] = sum;
  }
}

// ---------------------------------------------------------------------------
extern "C" void kernel_launch(void* const* d_in, const int* in_sizes, int n_in,
                              void* d_out, int out_size, void* d_ws, size_t ws_size,
                              hipStream_t stream) {
  const float* x   = (const float*)d_in[0];
  const float* W1  = (const float*)d_in[1];
  const float* b1  = (const float*)d_in[2];
  const float* g1  = (const float*)d_in[3];
  const float* be1 = (const float*)d_in[4];
  const float* m1  = (const float*)d_in[5];
  const float* v1  = (const float*)d_in[6];
  const float* W2  = (const float*)d_in[7];
  const float* b2  = (const float*)d_in[8];
  const float* g2  = (const float*)d_in[9];
  const float* be2 = (const float*)d_in[10];
  const float* m2  = (const float*)d_in[11];
  const float* v2  = (const float*)d_in[12];
  const float* W3  = (const float*)d_in[13];
  const float* b3  = (const float*)d_in[14];

  char* ws = (char*)d_ws;
  _Float16* w1frag = (_Float16*)(ws);              // 819,200 B
  _Float16* w2frag = (_Float16*)(ws + 819200);     //  32,768 B
  float*    th1    = (float*)(ws + 851968);        //   2,048 B
  float*    th2    = (float*)(ws + 854016);        //     128 B
  float*    w3b    = (float*)(ws + 854144);        //   1,280 B

  const int B = in_sizes[0] / 784;                 // 131072 (multiple of 64)

  bnn_prep<<<(426848 + 255) / 256, 256, 0, stream>>>(
      W1, b1, g1, be1, m1, v1, W2, b2, g2, be2, m2, v2, W3,
      w1frag, w2frag, th1, th2, w3b);

  bnn_fused<<<B / 64, 256, 0, stream>>>(
      x, w1frag, w2frag, th1, th2, w3b, b3, (float*)d_out);
}
